// Attention_76459007803452
// MI455X (gfx1250) — matrix operand order repaired and checked by reading, not compiled, not run on hardware
//
#include <hip/hip_runtime.h>

typedef __attribute__((ext_vector_type(16))) _Float16 v16h;
typedef __attribute__((ext_vector_type(8)))  _Float16 v8h;
typedef __attribute__((ext_vector_type(8)))  float    v8f;

#define DIM   1024
#define HEADS 16
#define DHEAD 64
#define BATCH 8
#define SEQ   1024
#define ROWS  (BATCH*SEQ)      // 8192
#define BH    (BATCH*HEADS)    // 128

// Async global->LDS copy (CDNA5, ASYNCcnt-tracked). VDST = LDS byte address,
// VADDR = 64-bit global address (ISA 15.18.3 opcode 98).
#define ASYNC_B128(ldsoff, gptr)                                           \
  asm volatile("global_load_async_to_lds_b128 %0, %1, off"                 \
               :: "v"(ldsoff), "v"((unsigned long long)(size_t)(gptr))     \
               : "memory")
#define WAIT_ASYNC0() asm volatile("s_wait_asynccnt 0" ::: "memory")

// ---------------------------------------------------------------------------
// LayerNorm: x[8192,1024] f32 -> xn f16, one 256-thread block per row
// ---------------------------------------------------------------------------
__global__ void __launch_bounds__(256) ln_kernel(const float* __restrict__ x,
                                                 const float* __restrict__ gamma,
                                                 _Float16* __restrict__ xn) {
  __shared__ float red[256];
  __shared__ float red2[256];
  int row = blockIdx.x;
  int t = threadIdx.x;
  const float* xr = x + (size_t)row * DIM;
  float v0 = xr[t], v1 = xr[t + 256], v2 = xr[t + 512], v3 = xr[t + 768];
  red[t]  = v0 + v1 + v2 + v3;
  red2[t] = v0 * v0 + v1 * v1 + v2 * v2 + v3 * v3;
  __syncthreads();
  for (int o = 128; o > 0; o >>= 1) {
    if (t < o) { red[t] += red[t + o]; red2[t] += red2[t + o]; }
    __syncthreads();
  }
  float mu  = red[0] * (1.0f / DIM);
  float var = red2[0] * (1.0f / DIM) - mu * mu;
  float rs  = rsqrtf(var + 1e-5f);
  _Float16* orow = xn + (size_t)row * DIM;
  orow[t]       = (_Float16)((v0 - mu) * rs * gamma[t]);
  orow[t + 256] = (_Float16)((v1 - mu) * rs * gamma[t + 256]);
  orow[t + 512] = (_Float16)((v2 - mu) * rs * gamma[t + 512]);
  orow[t + 768] = (_Float16)((v3 - mu) * rs * gamma[t + 768]);
}

// ---------------------------------------------------------------------------
// Tiled transpose+cast: in[rows][cols] f32 -> out[cols][rows] f16
// ---------------------------------------------------------------------------
__global__ void __launch_bounds__(256) transpose_f16(const float* __restrict__ in,
                                                     _Float16* __restrict__ out,
                                                     int rows, int cols) {
  __shared__ float tile[32][33];
  int c0 = blockIdx.x * 32, r0 = blockIdx.y * 32;
  int tx = threadIdx.x, ty = threadIdx.y;
#pragma unroll
  for (int i = 0; i < 32; i += 8)
    tile[ty + i][tx] = in[(size_t)(r0 + ty + i) * cols + (c0 + tx)];
  __syncthreads();
#pragma unroll
  for (int i = 0; i < 32; i += 8)
    out[(size_t)(c0 + ty + i) * rows + (r0 + tx)] = (_Float16)tile[tx][ty + i];
}

// ---------------------------------------------------------------------------
// Register-blocked 2x4 WMMA mainloop: one wave computes a 32x64 f32 block.
// 12 b128 loads feed 8 v_wmma per 32-deep K step (1.5 loads/WMMA).
// A row-major [M,K]; BT row-major [N,K] so B-frags are contiguous 32B loads.
// ---------------------------------------------------------------------------
__device__ __forceinline__ void wmma_block_2x4(const _Float16* __restrict__ A,
                                               const _Float16* __restrict__ BT,
                                               int lda, int ldb, int K,
                                               int row0, int col0,
                                               v8f (&acc)[2][4]) {
  int lane = threadIdx.x & 31;
  int h  = lane >> 4;
  int rc = lane & 15;
  const _Float16* ap[2];
  const _Float16* bp[4];
#pragma unroll
  for (int mi = 0; mi < 2; ++mi) ap[mi] = A + (size_t)(row0 + mi * 16 + rc) * lda;
#pragma unroll
  for (int ni = 0; ni < 4; ++ni) bp[ni] = BT + (size_t)(col0 + ni * 16 + rc) * ldb;
#pragma unroll
  for (int mi = 0; mi < 2; ++mi)
#pragma unroll
    for (int ni = 0; ni < 4; ++ni)
#pragma unroll
      for (int i = 0; i < 8; ++i) acc[mi][ni][i] = 0.0f;

  for (int k0 = 0; k0 < K; k0 += 32) {
    v16h a[2], b[4];
#pragma unroll
    for (int mi = 0; mi < 2; ++mi) {
      v8h lo, hi;
      __builtin_memcpy(&lo, ap[mi] + k0 + 16 * h, 16);
      __builtin_memcpy(&hi, ap[mi] + k0 + 16 * h + 8, 16);
#pragma unroll
      for (int i = 0; i < 8; ++i) { a[mi][i] = lo[i]; a[mi][8 + i] = hi[i]; }
      __builtin_prefetch(ap[mi] + k0 + 256, 0, 3);   // WGP-scope prefetch
    }
#pragma unroll
    for (int ni = 0; ni < 4; ++ni) {
      __builtin_memcpy(&b[ni], bp[ni] + k0 + 16 * h, 32);
      __builtin_prefetch(bp[ni] + k0 + 256, 0, 3);
    }
#pragma unroll
    for (int mi = 0; mi < 2; ++mi)
#pragma unroll
      for (int ni = 0; ni < 4; ++ni)
        acc[mi][ni] = __builtin_amdgcn_wmma_f32_16x16x32_f16(
            false, a[mi], false, b[ni], (short)0, acc[mi][ni], false, false);
  }
}

// ---------------------------------------------------------------------------
// QKV GEMM: xn[8192,1024] x wqkvT[3072,1024]; 8 waves (2x4) per block cover
// 64x256 output. Epilogue scatters f16 into q,k [bh][n][64] and vT [bh][64][n].
// Segment / head / batch selection is wave-uniform per tile (tiles are
// 16-aligned; segment and head boundaries are 1024/64-aligned), so branches
// are scalar (s_cbranch), no exec masking.
// ---------------------------------------------------------------------------
__global__ void __launch_bounds__(256)
gemm_qkv(const _Float16* __restrict__ xn, const _Float16* __restrict__ wT,
         _Float16* __restrict__ q, _Float16* __restrict__ kk,
         _Float16* __restrict__ vT) {
  int wid = threadIdx.x >> 5, lane = threadIdx.x & 31;
  int row0 = blockIdx.y * 64 + (wid >> 2) * 32;
  int col0 = blockIdx.x * 256 + (wid & 3) * 64;
  v8f acc[2][4];
  wmma_block_2x4(xn, wT, DIM, DIM, DIM, row0, col0, acc);
  int h = lane >> 4, cl = lane & 15;
#pragma unroll
  for (int mi = 0; mi < 2; ++mi) {
    int rbase = row0 + mi * 16;            // uniform
    int b = rbase >> 10;                   // uniform (tiles never cross batch)
    int nbase = (rbase & 1023) + 8 * h;    // + v per element
#pragma unroll
    for (int ni = 0; ni < 4; ++ni) {
      int tilec = col0 + ni * 16;          // uniform
      int seg = tilec >> 10;               // 0=q, 1=k, 2=v  (uniform)
      int c2 = tilec & 1023;
      int hh = c2 >> 6;                    // head (uniform; tile within head)
      int d = (c2 & 63) + cl;              // per-lane d within head
      size_t bh16 = (size_t)(b * 16 + hh);
      if (seg == 0) {
        _Float16* dst = q + ((bh16 * 1024 + nbase) << 6) + d;
#pragma unroll
        for (int v = 0; v < 8; ++v) dst[(size_t)v << 6] = (_Float16)acc[mi][ni][v];
      } else if (seg == 1) {
        _Float16* dst = kk + ((bh16 * 1024 + nbase) << 6) + d;
#pragma unroll
        for (int v = 0; v < 8; ++v) dst[(size_t)v << 6] = (_Float16)acc[mi][ni][v];
      } else {
        _Float16* dst = vT + ((bh16 << 6) + d) * 1024 + nbase;
#pragma unroll
        for (int v = 0; v < 8; ++v) dst[v] = (_Float16)acc[mi][ni][v];
      }
    }
  }
}

// ---------------------------------------------------------------------------
// RMS-norm in place over d=64: t = t/||t|| * 8 * gamma[h][d]
// ---------------------------------------------------------------------------
__global__ void __launch_bounds__(256) rms_kernel(_Float16* __restrict__ t,
                                                  const float* __restrict__ gamma) {
  int row = blockIdx.x * blockDim.x + threadIdx.x;
  if (row >= BH * SEQ) return;
  int h = (row >> 10) & 15;
  _Float16* p = t + (size_t)row * DHEAD;
  float vals[DHEAD];
  float ss = 0.0f;
#pragma unroll
  for (int d = 0; d < DHEAD; ++d) { vals[d] = (float)p[d]; ss += vals[d] * vals[d]; }
  float inv = 8.0f / fmaxf(sqrtf(ss), 1e-12f);
#pragma unroll
  for (int d = 0; d < DHEAD; ++d)
    p[d] = (_Float16)(vals[d] * inv * gamma[h * 64 + d]);
}

// ---------------------------------------------------------------------------
// Flash attention per (b,h): 4 waves x 16 query rows. K/V blocks (32 keys)
// are DMA'd into LDS once per workgroup with global_load_async_to_lds_b128,
// double-buffered so the next block's async copy overlaps current compute.
// Per block: 4 WMMAs for S, shfl row-reduce online softmax, exp->Pbuf->A-frag,
// 4 WMMAs accumulate O over d=64.
// ---------------------------------------------------------------------------
__global__ void __launch_bounds__(128)
attn_kernel(const _Float16* __restrict__ q, const _Float16* __restrict__ k,
            const _Float16* __restrict__ vT, _Float16* __restrict__ o16) {
  __shared__ __align__(16) _Float16 Kbuf[2][32][64];   // [key][d]
  __shared__ __align__(16) _Float16 Vbuf[2][64][32];   // [d][key]
  __shared__ __align__(16) _Float16 Pbuf[4][16][48];
  int tid = threadIdx.x;
  int w = tid >> 5, lane = tid & 31;
  int h = lane >> 4, cl = lane & 15;
  int bh = blockIdx.y;
  int q0 = blockIdx.x * 64 + w * 16;

  const _Float16* qbase = q + ((size_t)bh * SEQ + q0) * DHEAD;
  const _Float16* kbase = k + (size_t)bh * SEQ * DHEAD;
  const _Float16* vbase = vT + (size_t)bh * DHEAD * SEQ;

  unsigned int klds0 = (unsigned int)(size_t)(&Kbuf[0][0][0]);
  unsigned int vlds0 = (unsigned int)(size_t)(&Vbuf[0][0][0]);

  // Cooperative async stage of one 32-key block (8 KB) into buffer bi:
  // K block is 4KB contiguous; V block is 64 rows x 64B (stride 2KB).
  auto issue = [&](int bi, int j0) {
    const char* kg = (const char*)(kbase + (size_t)j0 * DHEAD) + tid * 32;
    unsigned int kd = klds0 + (unsigned int)(bi * 4096 + tid * 32);
    ASYNC_B128(kd, kg);
    ASYNC_B128(kd + 16, kg + 16);
    int d = tid >> 1, half = (tid & 1) * 32;
    const char* vg = (const char*)(vbase + (size_t)d * SEQ + j0) + half;
    unsigned int vd = vlds0 + (unsigned int)(bi * 4096 + d * 64 + half);
    ASYNC_B128(vd, vg);
    ASYNC_B128(vd + 16, vg + 16);
  };

  // Q A-frags for K-steps 0 and 32, resident in VGPRs
  v16h qa[2];
#pragma unroll
  for (int kk0 = 0; kk0 < 2; ++kk0) {
    const _Float16* ap = qbase + (size_t)cl * DHEAD + kk0 * 32;
    v8h lo, hi;
    __builtin_memcpy(&lo, ap + 16 * h, 16);
    __builtin_memcpy(&hi, ap + 16 * h + 8, 16);
#pragma unroll
    for (int i = 0; i < 8; ++i) { qa[kk0][i] = lo[i]; qa[kk0][8 + i] = hi[i]; }
  }

  float m[8], l[8];
  v8f oacc[4];
#pragma unroll
  for (int v = 0; v < 8; ++v) { m[v] = -__builtin_inff(); l[v] = 0.0f; }
#pragma unroll
  for (int t4 = 0; t4 < 4; ++t4)
#pragma unroll
    for (int v = 0; v < 8; ++v) oacc[t4][v] = 0.0f;

  issue(0, 0);

  for (int j0 = 0; j0 < SEQ; j0 += 32) {
    int cur = (j0 >> 5) & 1;
    WAIT_ASYNC0();        // our own async copies into buf[cur] are done
    __syncthreads();      // everyone's copies done; prev compute on buf[cur] done
    if (j0 + 32 < SEQ) issue(cur ^ 1, j0 + 32);

    // S = Q x K^T for 32 keys (two 16-col tiles), frags from LDS
    v8f s0, s1;
#pragma unroll
    for (int i = 0; i < 8; ++i) { s0[i] = 0.0f; s1[i] = 0.0f; }
#pragma unroll
    for (int kk0 = 0; kk0 < 2; ++kk0) {
      v16h b0, b1;  // b[i] = K[j0+col][kk0*32+16h+i]
      __builtin_memcpy(&b0, &Kbuf[cur][cl][kk0 * 32 + 16 * h], 32);
      __builtin_memcpy(&b1, &Kbuf[cur][16 + cl][kk0 * 32 + 16 * h], 32);
      s0 = __builtin_amdgcn_wmma_f32_16x16x32_f16(false, qa[kk0], false, b0,
                                                  (short)0, s0, false, false);
      s1 = __builtin_amdgcn_wmma_f32_16x16x32_f16(false, qa[kk0], false, b1,
                                                  (short)0, s1, false, false);
    }
    // online softmax per owned row r = 8h+v (row lives in 16 lanes of one half)
#pragma unroll
    for (int v = 0; v < 8; ++v) {
      float tmax = fmaxf(s0[v], s1[v]);
#pragma unroll
      for (int off = 1; off < 16; off <<= 1)
        tmax = fmaxf(tmax, __shfl_xor(tmax, off, 32));
      float mn = fmaxf(m[v], tmax);
      float sc = __expf(m[v] - mn);
      float p0 = __expf(s0[v] - mn);
      float p1 = __expf(s1[v] - mn);
      float su = p0 + p1;
#pragma unroll
      for (int off = 1; off < 16; off <<= 1)
        su += __shfl_xor(su, off, 32);
      l[v] = l[v] * sc + su;
      m[v] = mn;
#pragma unroll
      for (int t4 = 0; t4 < 4; ++t4) oacc[t4][v] *= sc;
      int r = 8 * h + v;
      Pbuf[w][r][cl]      = (_Float16)p0;
      Pbuf[w][r][16 + cl] = (_Float16)p1;
    }
    // reload P tile as an A-fragment (per-wave buffer, DS in-order per wave)
    v16h pa;
    {
      v8h lo, hi;
      __builtin_memcpy(&lo, &Pbuf[w][cl][16 * h], 16);
      __builtin_memcpy(&hi, &Pbuf[w][cl][16 * h + 8], 16);
#pragma unroll
      for (int i = 0; i < 8; ++i) { pa[i] = lo[i]; pa[8 + i] = hi[i]; }
    }
    // O += P(16x32) x V(32x64), V frags from LDS
#pragma unroll
    for (int t4 = 0; t4 < 4; ++t4) {
      v16h vb;  // b[i] = V[j0+16h+i][16t4+col] = Vbuf[16t4+col][16h+i]
      __builtin_memcpy(&vb, &Vbuf[cur][16 * t4 + cl][16 * h], 32);
      oacc[t4] = __builtin_amdgcn_wmma_f32_16x16x32_f16(false, pa, false, vb,
                                                        (short)0, oacc[t4], false, false);
    }
  }

  // finalize: divide by l, store f16 into [b, n, h*64+d] for the out-proj GEMM
  int b = bh >> 4, hh = bh & 15;
#pragma unroll
  for (int v = 0; v < 8; ++v) {
    float inv = 1.0f / l[v];
    int n = q0 + 8 * h + v;
#pragma unroll
    for (int t4 = 0; t4 < 4; ++t4) {
      int c = hh * 64 + 16 * t4 + cl;
      o16[(size_t)(b * SEQ + n) * DIM + c] = (_Float16)(oacc[t4][v] * inv);
    }
  }
}

// ---------------------------------------------------------------------------
// Output GEMM: o16[8192,1024] x woutT[1024,1024] -> f32 d_out
// ---------------------------------------------------------------------------
__global__ void __launch_bounds__(256)
gemm_out(const _Float16* __restrict__ A, const _Float16* __restrict__ BT,
         float* __restrict__ out) {
  int wid = threadIdx.x >> 5, lane = threadIdx.x & 31;
  int row0 = blockIdx.y * 64 + (wid >> 2) * 32;
  int col0 = blockIdx.x * 256 + (wid & 3) * 64;
  v8f acc[2][4];
  wmma_block_2x4(A, BT, DIM, DIM, DIM, row0, col0, acc);
  int h = lane >> 4, cl = lane & 15;
#pragma unroll
  for (int mi = 0; mi < 2; ++mi)
#pragma unroll
    for (int ni = 0; ni < 4; ++ni) {
      int c = col0 + ni * 16 + cl;
#pragma unroll
      for (int v = 0; v < 8; ++v)
        out[(size_t)(row0 + mi * 16 + 8 * h + v) * DIM + c] = acc[mi][ni][v];
    }
}

// ---------------------------------------------------------------------------
extern "C" void kernel_launch(void* const* d_in, const int* in_sizes, int n_in,
                              void* d_out, int out_size, void* d_ws, size_t ws_size,
                              hipStream_t stream) {
  const float* x     = (const float*)d_in[0];
  const float* ln_g  = (const float*)d_in[1];
  const float* q_g   = (const float*)d_in[2];
  const float* k_g   = (const float*)d_in[3];
  const float* w_qkv = (const float*)d_in[4];
  const float* w_out = (const float*)d_in[5];
  float* out = (float*)d_out;

  char* ws = (char*)d_ws;
  size_t off = 0;
  _Float16* xn    = (_Float16*)(ws + off); off += (size_t)ROWS * DIM * 2;        // 16 MB
  _Float16* wqkvT = (_Float16*)(ws + off); off += (size_t)3 * DIM * DIM * 2;     //  6 MB
  _Float16* woutT = (_Float16*)(ws + off); off += (size_t)DIM * DIM * 2;         //  2 MB
  _Float16* qf    = (_Float16*)(ws + off); off += (size_t)BH * SEQ * DHEAD * 2;  // 16 MB
  _Float16* kf    = (_Float16*)(ws + off); off += (size_t)BH * SEQ * DHEAD * 2;  // 16 MB
  _Float16* vTf   = (_Float16*)(ws + off); off += (size_t)BH * DHEAD * SEQ * 2;  // 16 MB
  _Float16* of    = (_Float16*)(ws + off); off += (size_t)ROWS * DIM * 2;        // 16 MB

  ln_kernel<<<ROWS, 256, 0, stream>>>(x, ln_g, xn);
  transpose_f16<<<dim3(3 * DIM / 32, DIM / 32), dim3(32, 8), 0, stream>>>(w_qkv, wqkvT, DIM, 3 * DIM);
  transpose_f16<<<dim3(DIM / 32, DIM / 32), dim3(32, 8), 0, stream>>>(w_out, woutT, DIM, DIM);
  gemm_qkv<<<dim3(3 * DIM / 256, ROWS / 64), 256, 0, stream>>>(xn, wqkvT, qf, kf, vTf);
  rms_kernel<<<(BH * SEQ + 255) / 256, 256, 0, stream>>>(qf, q_g);
  rms_kernel<<<(BH * SEQ + 255) / 256, 256, 0, stream>>>(kf, k_g);
  attn_kernel<<<dim3(SEQ / 64, BH), 128, 0, stream>>>(qf, kf, vTf, of);
  gemm_out<<<dim3(DIM / 256, ROWS / 64), 256, 0, stream>>>(of, woutT, out);
}
